// Attention_50405736186248
// MI455X (gfx1250) — compile-verified, matrix operand order checked by reading, not run on hardware
//
#include <hip/hip_runtime.h>

// ---------------------------------------------------------------------------
// Attention block for MI455X (gfx1250, wave32, WMMA + TDM).
// vh = [v_mid, 0, 0] => heads 4..11 contribute nothing; only heads 0..3
// (512 dims) are computed. bf16 WMMA everywhere (f32 accumulate), fp32 tile
// staging via double-buffered Tensor Data Mover DMA, causal flash attention.
// ---------------------------------------------------------------------------

#define S_LEN 2048
#define NH    4        // effective heads
#define HD    128
#define DQ    512      // NH*HD
#define ROWS  4096     // B*S
#define LSTR  33       // LDS row stride in floats (TDM pad: +1 dword / 32)

typedef __attribute__((ext_vector_type(16))) __bf16 v16bf;
typedef __attribute__((ext_vector_type(8)))  float  v8f;
typedef __attribute__((ext_vector_type(4)))  unsigned int u32x4;
typedef __attribute__((ext_vector_type(8)))  int          i32x8;
typedef __attribute__((ext_vector_type(4)))  int          i32x4;

union Frag16 { v16bf v; unsigned int w[8]; };

// Native float->bf16 (RNE); lowers to v_cvt_*bf16* on gfx1250.
__device__ __forceinline__ unsigned short f2bf(float f) {
  union { __bf16 b; unsigned short u; } r;
  r.b = (__bf16)f;
  return r.u;
}

__device__ __forceinline__ unsigned int packbf(float x0, float x1) {
  union { __bf16 b[2]; unsigned int u; } r;
  r.b[0] = (__bf16)x0;
  r.b[1] = (__bf16)x1;
  return r.u;
}

// 16-bit A/B fragment K mapping (ISA 7.12.2): lane half 'hi', dword j holds
// K = {8*hi+2j, +1} for j<4 and K = {16+8*hi+2(j-4), +1} for j>=4.
__device__ __forceinline__ int kpair(int j, int hi) {
  return (j < 4) ? (8 * hi + 2 * j) : (16 + 8 * hi + 2 * (j - 4));
}

__device__ __forceinline__ v8f vzero() {
  v8f z;
#pragma unroll
  for (int i = 0; i < 8; ++i) z[i] = 0.0f;
  return z;
}

__device__ __forceinline__ v8f wmma_bf16(const Frag16& a, const Frag16& b, v8f c) {
  return __builtin_amdgcn_wmma_f32_16x16x32_bf16(
      /*neg_a=*/false, a.v, /*neg_b=*/false, b.v,
      /*c_mod=*/(short)0, c, /*reuse_a=*/false, /*reuse_b=*/false);
}

// ---------------------------------------------------------------------------
// TDM: 2D fp32 tile (rows x cols, row stride = stride_elems) -> LDS, with
// LDS padding of 1 dword after every 32 dwords (row stride 33 floats).
// D# packing per CDNA5 ISA:
//   g0: [1:0]=count=1 | [63:32]=lds_addr | [120:64]=global_addr | [127:126]=2
//   g1: [17:16]=data_size(2=>4B) | [20]=pad_en | [24:22]=pad_interval(4=>32dw)
//       [31:25]=pad_amount(0=>1dw) | [79:48]=tensor_dim0 | [111:80]=tensor_dim1
//       [127:112]=tile_dim0 | [143:128]=tile_dim1 | [207:160]=dim0_stride
// tensor dims == tile dims (tiles are always fully in-bounds here).
// ---------------------------------------------------------------------------
#if __has_builtin(__builtin_amdgcn_tensor_load_to_lds)
#define HAVE_TDM 1
__device__ __forceinline__ void tdm_load_2d_f32(
    unsigned lds_off, const float* gptr,
    unsigned rows, unsigned cols, unsigned stride_elems)
{
  const unsigned long long ga = (unsigned long long)(size_t)gptr;
  u32x4 g0;
  g0[0] = 1u;                                                 // count=1
  g0[1] = lds_off;                                            // LDS byte addr
  g0[2] = (unsigned)(ga & 0xFFFFFFFFu);                       // gaddr[31:0]
  g0[3] = (unsigned)((ga >> 32) & 0x01FFFFFFu) | (2u << 30);  // gaddr[56:32]|type=2
  i32x8 g1;
  g1[0] = (int)(0x00020000u | (1u << 20) | (4u << 22));       // 4B, pad 1dw/32dw
  g1[1] = (int)((cols & 0xFFFFu) << 16);                      // tdim0[15:0]
  g1[2] = (int)(((cols >> 16) & 0xFFFFu) | ((rows & 0xFFFFu) << 16)); // tdim0 hi|tdim1 lo
  g1[3] = (int)(((rows >> 16) & 0xFFFFu) | ((cols & 0xFFFFu) << 16)); // tdim1 hi|tile_dim0
  g1[4] = (int)(rows & 0xFFFFu);                              // tile_dim1 (tile_dim2=0)
  g1[5] = (int)stride_elems;                                  // dim0_stride[31:0]
  g1[6] = 0;                                                  // stride hi | dim1_stride lo
  g1[7] = 0;
  const i32x4 z4 = {0, 0, 0, 0};
#if __clang_major__ >= 23
  const i32x8 z8 = {0, 0, 0, 0, 0, 0, 0, 0};
  __builtin_amdgcn_tensor_load_to_lds(g0, g1, z4, z4, z8, 0);
#else
  __builtin_amdgcn_tensor_load_to_lds(g0, g1, z4, z4, 0);
#endif
}
#else
#define HAVE_TDM 0
#endif

// ---------------------------------------------------------------------------
// C[M,N] = A[M,K] * B[N,K]^T  (fp32 in/out, bf16 WMMA inside).
// Block: 128 threads = 4 waves, 64x64 output tile, K-step 32.
// Double-buffered: wave 0 issues next tile's TDM loads, then waits for the
// current tile (tensorcnt<=2; tensor ops complete in-order per wave).
// ---------------------------------------------------------------------------
__global__ __launch_bounds__(128) void gemm_abT(
    const float* __restrict__ A, int lda,
    const float* __restrict__ Bm, int ldb,
    float* __restrict__ C, int ldc, int K)
{
  __shared__ __align__(16) float sAf[2][64 * LSTR];
  __shared__ __align__(16) float sBf[2][64 * LSTR];
  const int tid  = threadIdx.x;
  const int wave = tid >> 5, lane = tid & 31;
  const int n16  = lane & 15, hi = lane >> 4;
  const int M0 = blockIdx.y * 64, N0 = blockIdx.x * 64;

  v8f acc[4];
#pragma unroll
  for (int nt = 0; nt < 4; ++nt) acc[nt] = vzero();

  const int nbuf = K >> 5;   // number of 32-wide K tiles

#if HAVE_TDM
  if (wave == 0) {
    tdm_load_2d_f32((unsigned)(size_t)&sAf[0][0], A + (size_t)M0 * lda,
                    64u, 32u, (unsigned)lda);
    tdm_load_2d_f32((unsigned)(size_t)&sBf[0][0], Bm + (size_t)N0 * ldb,
                    64u, 32u, (unsigned)ldb);
  }
#endif

  for (int it = 0; it < nbuf; ++it) {
    const int cur = it & 1;
#if HAVE_TDM
    if (wave == 0) {
      if (it + 1 < nbuf) {
        const int k1 = (it + 1) << 5;
        tdm_load_2d_f32((unsigned)(size_t)&sAf[cur ^ 1][0],
                        A + (size_t)M0 * lda + k1, 64u, 32u, (unsigned)lda);
        tdm_load_2d_f32((unsigned)(size_t)&sBf[cur ^ 1][0],
                        Bm + (size_t)N0 * ldb + k1, 64u, 32u, (unsigned)ldb);
        __builtin_amdgcn_s_wait_tensorcnt((short)2);   // current tile done
      } else {
        __builtin_amdgcn_s_wait_tensorcnt((short)0);
      }
    }
    __syncthreads();
#else
    {
      const int k0 = it << 5;
      const int lr = tid >> 3;         // 0..15
      const int lc = (tid & 7) << 2;   // 0,4,...,28
#pragma unroll
      for (int rr = 0; rr < 64; rr += 16) {
        const float4 a4 = *(const float4*)(A + (size_t)(M0 + lr + rr) * lda + k0 + lc);
        *(float4*)&sAf[cur][(lr + rr) * LSTR + lc] = a4;
        const float4 b4 = *(const float4*)(Bm + (size_t)(N0 + lr + rr) * ldb + k0 + lc);
        *(float4*)&sBf[cur][(lr + rr) * LSTR + lc] = b4;
      }
    }
    __syncthreads();
#endif

    Frag16 af;
#pragma unroll
    for (int j = 0; j < 8; ++j) {
      const int kp = kpair(j, hi);
      const float* p = &sAf[cur][(wave * 16 + n16) * LSTR + kp];
      af.w[j] = packbf(p[0], p[1]);
    }
#pragma unroll
    for (int nt = 0; nt < 4; ++nt) {
      Frag16 bf;
#pragma unroll
      for (int j = 0; j < 8; ++j) {
        const int kp = kpair(j, hi);
        const float* p = &sBf[cur][(nt * 16 + n16) * LSTR + kp];
        bf.w[j] = packbf(p[0], p[1]);
      }
      acc[nt] = wmma_bf16(af, bf, acc[nt]);
    }
    __syncthreads();   // buffer may be overwritten next iteration
  }

  // C/D layout: vgpr r -> row r + 8*hi, col = lane&15
#pragma unroll
  for (int nt = 0; nt < 4; ++nt)
#pragma unroll
    for (int r = 0; r < 8; ++r)
      C[(size_t)(M0 + wave * 16 + r + 8 * hi) * ldc + N0 + nt * 16 + n16] = acc[nt][r];
}

// ---------------------------------------------------------------------------
// Interleaved-pair RoPE (torchtune convention), fp32 -> bf16.
// ---------------------------------------------------------------------------
__global__ __launch_bounds__(256) void rope_bf16(
    const float* __restrict__ src, unsigned short* __restrict__ dst)
{
  const int idx = blockIdx.x * 256 + threadIdx.x;
  const int row = idx >> 8;                 // b*S + s
  const int p   = idx & 255;                // pair within 512-dim row
  const int s   = row & (S_LEN - 1);
  const int pp  = p & 63;                   // pair within head
  const int c0  = (p >> 6) * HD + 2 * pp;
  const float freq = __expf(-0.14391156f * (float)pp);  // 10000^(-pp/64)
  const float ang  = (float)s * freq;
  float sn, cs;
  __sincosf(ang, &sn, &cs);
  const float x1 = src[(size_t)row * DQ + c0];
  const float x2 = src[(size_t)row * DQ + c0 + 1];
  *(unsigned int*)(dst + (size_t)row * DQ + c0) =
      packbf(x1 * cs - x2 * sn, x1 * sn + x2 * cs);
}

// ---------------------------------------------------------------------------
// v_mid fp32 [b*S+s][h*HD+d] -> V^T bf16 [(b*NH+h)*HD+d][s]
// ---------------------------------------------------------------------------
__global__ __launch_bounds__(256) void vtrans(
    const float* __restrict__ vm, unsigned short* __restrict__ vt)
{
  const int idx = blockIdx.x * 256 + threadIdx.x;   // ROWS*DQ threads
  const int row = idx >> 9;
  const int c   = idx & 511;
  const int b   = row >> 11, s = row & (S_LEN - 1);
  vt[(size_t)(b * DQ + c) * S_LEN + s] = f2bf(vm[(size_t)row * DQ + c]);
}

// ---------------------------------------------------------------------------
// Causal flash attention, 4 heads, d=128. Block = 4 waves; each wave owns a
// 16-row Q tile and streams keys in 32-wide steps: 8 QK^T WMMAs + online
// softmax + 8 PV WMMAs per step. P is staged through wave-private LDS.
// ---------------------------------------------------------------------------
__global__ __launch_bounds__(128) void flash_attn(
    const unsigned short* __restrict__ qb,
    const unsigned short* __restrict__ kb,
    const unsigned short* __restrict__ vt,
    float* __restrict__ attn)
{
  __shared__ unsigned short ldsP[4][16 * 34];
  const int tid  = threadIdx.x;
  const int wave = tid >> 5, lane = tid & 31;
  const int n16  = lane & 15, hi = lane >> 4;
  const int bid  = blockIdx.x;               // [b][h][qblk]
  const int qblk = bid & 31;
  const int h    = (bid >> 5) & 3;
  const int b    = bid >> 7;
  const int q0    = qblk * 64 + wave * 16;
  const int bbase = b * S_LEN;
  const int hcol  = h * HD;
  const float scale = 0.08838834764831845f;  // 1/sqrt(128)

  // Q fragments: 4 x (16x32) covering d=128
  Frag16 qf[4];
#pragma unroll
  for (int f = 0; f < 4; ++f) {
    const size_t base = (size_t)(bbase + q0 + n16) * DQ + hcol + f * 32;
#pragma unroll
    for (int j = 0; j < 8; ++j)
      qf[f].w[j] = *(const unsigned int*)(qb + base + kpair(j, hi));
  }

  float run_m[8], run_l[8];
  v8f oacc[8];
#pragma unroll
  for (int r = 0; r < 8; ++r) { run_m[r] = -1e30f; run_l[r] = 0.0f; }
#pragma unroll
  for (int j8 = 0; j8 < 8; ++j8) oacc[j8] = vzero();

  unsigned short* Pw = ldsP[wave];

  for (int k0 = 0; k0 <= q0 + 15; k0 += 32) {
    // Prefetch next K tile rows for this lane (global_prefetch_b8).
    if (k0 + 32 + n16 <= q0 + 15) {
      __builtin_prefetch(kb + (size_t)(bbase + k0 + 32 + n16) * DQ + hcol, 0, 0);
      __builtin_prefetch(kb + (size_t)(bbase + k0 + 48 + n16) * DQ + hcol, 0, 0);
    }

    // ---- scores: two 16-col tiles (keys k0..k0+15, k0+16..k0+31) ----
    v8f c0 = vzero(), c1 = vzero();
#pragma unroll
    for (int f = 0; f < 4; ++f) {
      Frag16 bk0, bk1;
      int key0 = k0 + n16;      if (key0 > S_LEN - 1) key0 = S_LEN - 1;   // masked anyway
      int key1 = k0 + 16 + n16; if (key1 > S_LEN - 1) key1 = S_LEN - 1;
      const size_t base0 = (size_t)(bbase + key0) * DQ + hcol + f * 32;
      const size_t base1 = (size_t)(bbase + key1) * DQ + hcol + f * 32;
#pragma unroll
      for (int j = 0; j < 8; ++j) {
        bk0.w[j] = *(const unsigned int*)(kb + base0 + kpair(j, hi));
        bk1.w[j] = *(const unsigned int*)(kb + base1 + kpair(j, hi));
      }
      c0 = wmma_bf16(qf[f], bk0, c0);
      c1 = wmma_bf16(qf[f], bk1, c1);
    }

    // ---- online softmax; each row lives in one 16-lane half ----
#pragma unroll
    for (int r = 0; r < 8; ++r) {
      const int row  = q0 + r + 8 * hi;
      const int col0 = k0 + n16, col1 = col0 + 16;
      float s0 = (col0 <= row) ? c0[r] * scale : -1e30f;
      float s1 = (col1 <= row) ? c1[r] * scale : -1e30f;
      float tmax = fmaxf(s0, s1);
      tmax = fmaxf(tmax, __shfl_xor(tmax, 1, 32));
      tmax = fmaxf(tmax, __shfl_xor(tmax, 2, 32));
      tmax = fmaxf(tmax, __shfl_xor(tmax, 4, 32));
      tmax = fmaxf(tmax, __shfl_xor(tmax, 8, 32));
      const float nm   = fmaxf(run_m[r], tmax);
      const float corr = __expf(run_m[r] - nm);
      const float p0   = __expf(s0 - nm);
      const float p1   = __expf(s1 - nm);
      float rs = p0 + p1;
      rs += __shfl_xor(rs, 1, 32);
      rs += __shfl_xor(rs, 2, 32);
      rs += __shfl_xor(rs, 4, 32);
      rs += __shfl_xor(rs, 8, 32);
      run_l[r] = run_l[r] * corr + rs;
      run_m[r] = nm;
#pragma unroll
      for (int j8 = 0; j8 < 8; ++j8) oacc[j8][r] *= corr;
      const int prow = r + 8 * hi;
      Pw[prow * 34 + n16]      = f2bf(p0);
      Pw[prow * 34 + 16 + n16] = f2bf(p1);
    }

    // DS ops are in-order within a wave; fence compiler + HW before re-read.
    asm volatile("s_wait_dscnt 0" ::: "memory");

    // ---- re-gather P as A fragment (16x32), then PV WMMAs ----
    Frag16 pf;
#pragma unroll
    for (int j = 0; j < 8; ++j)
      pf.w[j] = *(const unsigned int*)&Pw[n16 * 34 + kpair(j, hi)];
#pragma unroll
    for (int j8 = 0; j8 < 8; ++j8) {
      Frag16 vf;
      const size_t vbase = (size_t)((b * NH + h) * HD + j8 * 16 + n16) * S_LEN;
#pragma unroll
      for (int j = 0; j < 8; ++j) {
        int kk = k0 + kpair(j, hi);
        if (kk > S_LEN - 2) kk = S_LEN - 2;   // clamp: p=0 there, data finite
        vf.w[j] = *(const unsigned int*)(vt + vbase + kk);
      }
      oacc[j8] = wmma_bf16(pf, vf, oacc[j8]);
    }
  }

  // ---- normalize + store fp32 [ROWS x DQ] ----
#pragma unroll
  for (int j8 = 0; j8 < 8; ++j8)
#pragma unroll
    for (int r = 0; r < 8; ++r)
      attn[(size_t)(bbase + q0 + r + 8 * hi) * DQ + hcol + j8 * 16 + n16] =
          oacc[j8][r] / run_l[r];
}

// ---------------------------------------------------------------------------
extern "C" void kernel_launch(void* const* d_in, const int* in_sizes, int n_in,
                              void* d_out, int out_size, void* d_ws, size_t ws_size,
                              hipStream_t stream) {
  const float* q  = (const float*)d_in[0];
  const float* k  = (const float*)d_in[1];
  const float* v  = (const float*)d_in[2];
  const float* Wq = (const float*)d_in[3];
  const float* Wk = (const float*)d_in[4];
  const float* Wv = (const float*)d_in[5];
  const float* Wo = (const float*)d_in[6];
  float* out = (float*)d_out;

  // Workspace layout (44 MiB total, no overreads):
  char* w = (char*)d_ws;
  const size_t MB = 1u << 20;
  float*          qh = (float*)(w + 0 * MB);           // [4096,512] fp32
  float*          kh = (float*)(w + 8 * MB);           // [4096,512] fp32
  float*          vm = (float*)(w + 16 * MB);          // [4096,512] fp32
  float*          at = (float*)(w + 24 * MB);          // [4096,512] fp32
  unsigned short* qb = (unsigned short*)(w + 32 * MB); // [4096,512] bf16
  unsigned short* kb = (unsigned short*)(w + 36 * MB); // [4096,512] bf16
  unsigned short* vt = (unsigned short*)(w + 40 * MB); // [8,128,2048] bf16 (V^T)

  const dim3 blk(128);
  // qh = q @ Wq[0:512,:]^T   (M=4096, N=512, K=1536)
  gemm_abT<<<dim3(8, 64), blk, 0, stream>>>(q, 1536, Wq, 1536, qh, 512, 1536);
  // kh = k @ Wk[0:512,:]^T
  gemm_abT<<<dim3(8, 64), blk, 0, stream>>>(k, 1536, Wk, 1536, kh, 512, 1536);
  // v_mid = v[:,:,512:1024] @ Wv^T   (M=4096, N=512, K=512)
  gemm_abT<<<dim3(8, 64), blk, 0, stream>>>(v + 512, 1536, Wv, 512, vm, 512, 512);

  rope_bf16<<<4096, 256, 0, stream>>>(qh, qb);
  rope_bf16<<<4096, 256, 0, stream>>>(kh, kb);
  vtrans<<<8192, 256, 0, stream>>>(vm, vt);

  flash_attn<<<256, blk, 0, stream>>>(qb, kb, vt, at);

  // out = attn512 @ Wo[:, 0:512]^T   (M=4096, N=1536, K=512)
  gemm_abT<<<dim3(24, 64), blk, 0, stream>>>(at, 512, Wo, 1536, out, 1536, 512);
}
